// VTN_64201171140851
// MI455X (gfx1250) — matrix operand order repaired and checked by reading, not compiled, not run on hardware
//
#include <hip/hip_runtime.h>
#include <hip/hip_bf16.h>

// ---------------- model constants ----------------
#define B_      8
#define F_      255
#define E_      768
#define NH_     12
#define HD_     64
#define NL_     3
#define S_      256          // 1 cls + 255 tokens
#define MAXPOS_ 288
#define INTER_  3072
#define MLPD_   768
#define NCLS_   400
#define WHALF_  16           // window half: scores over 33 keys + 1 global

typedef __bf16 bf16;
typedef bf16  v8bf  __attribute__((ext_vector_type(8)));
typedef bf16  v16bf __attribute__((ext_vector_type(16)));
typedef float v8f   __attribute__((ext_vector_type(8)));

// ---------------- small device helpers ----------------
__device__ __forceinline__ unsigned short f2bf_bits(float f) {
    unsigned int u = __float_as_uint(f);
    unsigned int r = u + 0x7FFFu + ((u >> 16) & 1u);   // round-to-nearest-even
    return (unsigned short)(r >> 16);
}
__device__ __forceinline__ float gelu_exact(float x) {
    return 0.5f * x * (1.0f + erff(x * 0.70710678118654752f));
}
__device__ __forceinline__ float blockSum(float v) {
    __shared__ float buf[256];
    int t = threadIdx.x;
    buf[t] = v; __syncthreads();
    for (int s = blockDim.x >> 1; s > 0; s >>= 1) {
        if (t < s) buf[t] += buf[t + s];
        __syncthreads();
    }
    float r = buf[0]; __syncthreads();
    return r;
}
__device__ __forceinline__ float blockMax(float v) {
    __shared__ float buf[256];
    int t = threadIdx.x;
    buf[t] = v; __syncthreads();
    for (int s = blockDim.x >> 1; s > 0; s >>= 1) {
        if (t < s) buf[t] = fmaxf(buf[t], buf[t + s]);
        __syncthreads();
    }
    float r = buf[0]; __syncthreads();
    return r;
}
// CDNA5 async global->LDS copy of 16 bytes per lane (tracked by ASYNCcnt)
__device__ __forceinline__ void async_b128(unsigned int lds_off, const void* gaddr) {
    asm volatile("global_load_async_to_lds_b128 %0, %1, off"
                 :: "v"(lds_off), "v"(gaddr) : "memory");
}

// ---------------- f32 -> bf16 convert ----------------
__global__ void lf_cvt(const float* __restrict__ src, unsigned short* __restrict__ dst, size_t n) {
    size_t i = (size_t)blockIdx.x * blockDim.x + threadIdx.x;
    size_t stride = (size_t)gridDim.x * blockDim.x;
    for (; i < n; i += stride) dst[i] = f2bf_bits(src[i]);
}
// f32 [K,N] -> bf16 transposed [N,K] (weights), so GEMM B-tiles are contiguous copies
__global__ void lf_cvtT(const float* __restrict__ src, unsigned short* __restrict__ dst,
                        int K, int N) {
    size_t total = (size_t)K * N;
    size_t i = (size_t)blockIdx.x * blockDim.x + threadIdx.x;
    size_t stride = (size_t)gridDim.x * blockDim.x;
    for (; i < total; i += stride) {
        size_t k = i / N, n = i % N;
        dst[n * (size_t)K + k] = f2bf_bits(src[i]);
    }
}

// ---------------- stage 1: per-position LN over E then max over 7x7 ----------------
__global__ void lf_lnmax(const float* __restrict__ x, const float* __restrict__ g,
                         const float* __restrict__ bta, float* __restrict__ out) {
    int bf = blockIdx.x;               // b*F + f
    int t = threadIdx.x;               // 256 threads, 3 elems each
    const float* base = x + (size_t)bf * (49 * E_);
    float m0 = -1e30f, m1 = -1e30f, m2 = -1e30f;
    for (int p = 0; p < 49; ++p) {
        const float* row = base + p * E_;
        float v0 = row[t], v1 = row[t + 256], v2 = row[t + 512];
        float mean = blockSum(v0 + v1 + v2) * (1.0f / E_);
        float d0 = v0 - mean, d1 = v1 - mean, d2 = v2 - mean;
        float var = blockSum(d0 * d0 + d1 * d1 + d2 * d2) * (1.0f / E_);
        float inv = rsqrtf(var + 1e-6f);
        m0 = fmaxf(m0, d0 * inv * g[t]       + bta[t]);
        m1 = fmaxf(m1, d1 * inv * g[t + 256] + bta[t + 256]);
        m2 = fmaxf(m2, d2 * inv * g[t + 512] + bta[t + 512]);
    }
    float* o = out + (size_t)bf * E_;
    o[t] = m0; o[t + 256] = m1; o[t + 512] = m2;
}

// ---------------- stage 2: cls + pos + type embedding + LN ----------------
__global__ void lf_embed(const float* __restrict__ tok, const int* __restrict__ pos,
                         const float* __restrict__ cls, const float* __restrict__ pe,
                         const float* __restrict__ te, const float* __restrict__ g,
                         const float* __restrict__ bta, float* __restrict__ h) {
    int row = blockIdx.x;
    int b = row / S_, s = row % S_;
    int t = threadIdx.x;
    float v0, v1, v2;
    if (s == 0) {
        const float* pp = pe + (size_t)(MAXPOS_ - 2) * E_;
        v0 = cls[t]       + pp[t]       + te[E_ + t];
        v1 = cls[t + 256] + pp[t + 256] + te[E_ + t + 256];
        v2 = cls[t + 512] + pp[t + 512] + te[E_ + t + 512];
    } else {
        int pid = pos[b * F_ + (s - 1)] % (MAXPOS_ - 2);
        const float* xr = tok + ((size_t)b * F_ + (s - 1)) * E_;
        const float* pp = pe + (size_t)pid * E_;
        v0 = xr[t]       + pp[t]       + te[t];
        v1 = xr[t + 256] + pp[t + 256] + te[t + 256];
        v2 = xr[t + 512] + pp[t + 512] + te[t + 512];
    }
    float mean = blockSum(v0 + v1 + v2) * (1.0f / E_);
    float d0 = v0 - mean, d1 = v1 - mean, d2 = v2 - mean;
    float var = blockSum(d0 * d0 + d1 * d1 + d2 * d2) * (1.0f / E_);
    float inv = rsqrtf(var + 1e-5f);
    float* hr = h + (size_t)row * E_;
    hr[t]       = d0 * inv * g[t]       + bta[t];
    hr[t + 256] = d1 * inv * g[t + 256] + bta[t + 256];
    hr[t + 512] = d2 * inv * g[t + 512] + bta[t + 512];
}

// ---------------- WMMA GEMM ----------------
// C[M,N] = act((A_bf16[M,K] @ B, + bias) * alpha), with B given TRANSPOSED: BT[N,K].
// Async double-buffered global->LDS staging (GLOBAL_LOAD_ASYNC_TO_LDS_B128, ASYNCcnt).
#define LDAS 40   // LDS row stride (bf16): 80B -> 16B aligned, bank-conflict-free
#define LDBS 40
#define ATILE (128 * LDAS)
#define BTILE (64 * LDBS)

__device__ __forceinline__ void lf_stage(const bf16* __restrict__ A, int lda,
                                         const bf16* __restrict__ BT, int ldbt,
                                         int mBase, int nBase, int M, int k0,
                                         bf16* AsBuf, bf16* BsBuf, int t) {
    // A tile: 128 rows x 32 k = 512 16B-chunks, 2 per thread (rows clamped; unused rows masked at store)
    {
        int r = t >> 2, kc = (t & 3) << 3;
        int grow = mBase + r; if (grow >= M) grow = M - 1;
        async_b128((unsigned int)(size_t)(AsBuf + r * LDAS + kc),
                   (const void*)(A + (size_t)grow * lda + (k0 + kc)));
        int c1 = t + 256;
        r = c1 >> 2; kc = (c1 & 3) << 3;
        grow = mBase + r; if (grow >= M) grow = M - 1;
        async_b128((unsigned int)(size_t)(AsBuf + r * LDAS + kc),
                   (const void*)(A + (size_t)grow * lda + (k0 + kc)));
    }
    // BT tile: 64 rows (n) x 32 k = 256 16B-chunks, 1 per thread
    {
        int r = t >> 2, kc = (t & 3) << 3;
        async_b128((unsigned int)(size_t)(BsBuf + r * LDBS + kc),
                   (const void*)(BT + (size_t)(nBase + r) * ldbt + (k0 + kc)));
    }
}

__global__ __launch_bounds__(256) void lf_gemm(
    const bf16* __restrict__ A, int lda,
    const bf16* __restrict__ BT, int ldbt,
    void* __restrict__ Cout, int ldc,
    const float* __restrict__ bias,
    int M, int N, int K, float alpha, int act, int outbf) {
    __shared__ bf16 As[2 * ATILE];    // double-buffered 128x32 A tiles
    __shared__ bf16 Bs[2 * BTILE];    // double-buffered 64x32 B^T tiles
    const int t = threadIdx.x;
    const int lane = t & 31, wid = t >> 5;
    const int m0 = (wid >> 1) * 32;   // 4 waves along M
    const int n0 = (wid & 1) * 32;    // 2 waves along N
    const int lh = lane >> 4, ll = lane & 15;
    const int mBase = blockIdx.y * 128;
    const int nBase = blockIdx.x * 64;

    const v8f vz = {0.f, 0.f, 0.f, 0.f, 0.f, 0.f, 0.f, 0.f};
    v8f acc[2][2];
    acc[0][0] = vz; acc[0][1] = vz; acc[1][0] = vz; acc[1][1] = vz;

    const int nk = K >> 5;
    lf_stage(A, lda, BT, ldbt, mBase, nBase, M, 0, As, Bs, t);   // prefetch tile 0

    for (int kt = 0; kt < nk; ++kt) {
        const int cur = kt & 1;
        bf16* Ac = As + cur * ATILE;
        bf16* Bc = Bs + cur * BTILE;
        if (kt + 1 < nk) {
            // prefetch next tile into the other buffer (its last readers passed the
            // trailing barrier of iteration kt-1), then wait for current tile only:
            // async ops complete in order; 3 newest = prefetch, older = current tile.
            lf_stage(A, lda, BT, ldbt, mBase, nBase, M, (kt + 1) << 5,
                     As + (1 - cur) * ATILE, Bs + (1 - cur) * BTILE, t);
            asm volatile("s_wait_asynccnt 0x3" ::: "memory");
        } else {
            asm volatile("s_wait_asynccnt 0x0" ::: "memory");
        }
        __syncthreads();   // current tile visible to all waves

        // fragments per ISA 16-bit layouts:
        // A (16x32): lane ll = row; elems 0..7 -> K = lh*8+e ; elems 8..15 -> K = 16+lh*8+(e-8)
        // B (32x16): lane ll = col; elems 0..7 -> K = lh*8+e ; elems 8..15 -> K = 16+lh*8+(e-8)   (BT rows are N, contiguous K)
        v16bf af[2], bf2[2];
#pragma unroll
        for (int mt = 0; mt < 2; ++mt) {
            const bf16* pr = &Ac[(m0 + mt * 16 + ll) * LDAS];
            v8bf lo = *(const v8bf*)(pr + lh * 8);
            v8bf hi = *(const v8bf*)(pr + 16 + lh * 8);
            af[mt] = __builtin_shufflevector(lo, hi, 0, 1, 2, 3, 4, 5, 6, 7, 8, 9, 10, 11, 12, 13, 14, 15);
        }
#pragma unroll
        for (int nt = 0; nt < 2; ++nt) {
            const bf16* pc = &Bc[(n0 + nt * 16 + ll) * LDBS];
            v8bf lo = *(const v8bf*)(pc + lh * 8);
            v8bf hi = *(const v8bf*)(pc + 16 + lh * 8);
            bf2[nt] = __builtin_shufflevector(lo, hi, 0, 1, 2, 3, 4, 5, 6, 7, 8, 9, 10, 11, 12, 13, 14, 15);
        }
#pragma unroll
        for (int mt = 0; mt < 2; ++mt)
#pragma unroll
            for (int nt = 0; nt < 2; ++nt)
                acc[mt][nt] = __builtin_amdgcn_wmma_f32_16x16x32_bf16(
                    false, af[mt], false, bf2[nt], (short)0, acc[mt][nt], false, false);
        __syncthreads();   // all waves done reading cur before it is overwritten
    }

    // epilogue: C layout per ISA -> lane ll = N col, row = lh*8 + r
#pragma unroll
    for (int mt = 0; mt < 2; ++mt) {
#pragma unroll
        for (int nt = 0; nt < 2; ++nt) {
            int gn = nBase + n0 + nt * 16 + ll;
#pragma unroll
            for (int r = 0; r < 8; ++r) {
                int gm = mBase + m0 + mt * 16 + lh * 8 + r;
                if (gm < M) {
                    float vv = acc[mt][nt][r];
                    if (bias) vv += bias[gn];
                    vv *= alpha;
                    if (act) vv = gelu_exact(vv);
                    if (outbf)
                        ((unsigned short*)Cout)[(size_t)gm * ldc + gn] = f2bf_bits(vv);
                    else
                        ((float*)Cout)[(size_t)gm * ldc + gn] = vv;
                }
            }
        }
    }
}

// ---------------- sliding-window attention (33 keys + global col 0) ----------------
__global__ void lf_attn_local(const float* __restrict__ q, const float* __restrict__ k,
                              const float* __restrict__ v, float* __restrict__ o) {
    int s = blockIdx.x, hh = blockIdx.y, b = blockIdx.z;
    int t = threadIdx.x;                                    // 64 threads
    __shared__ float sc[34];
    const float* qp = q + ((size_t)b * S_ + s) * E_ + hh * HD_;
    if (t < 34) {
        int key; bool valid = true;
        if (t == 0) { key = 0; }                            // global column (k[:,0])
        else {
            int idx = s - WHALF_ + (t - 1);
            valid = (idx >= 0) && (idx < S_);
            key = min(max(idx, 0), S_ - 1);
        }
        const float* kp = k + ((size_t)b * S_ + key) * E_ + hh * HD_;
        float d = 0.f;
        for (int i = 0; i < HD_; ++i) d += qp[i] * kp[i];
        sc[t] = valid ? d : -1e9f;
    }
    __syncthreads();
    float mx = -1e30f;
    for (int i = 0; i < 34; ++i) mx = fmaxf(mx, sc[i]);
    float den = 0.f;
    for (int i = 0; i < 34; ++i) den += expf(sc[i] - mx);
    float inv = 1.0f / den;
    float accv = expf(sc[0] - mx) * inv * v[((size_t)b * S_) * E_ + hh * HD_ + t];
    for (int w = 0; w < 33; ++w) {
        int idx = s - WHALF_ + w;
        int key = min(max(idx, 0), S_ - 1);
        accv += expf(sc[1 + w] - mx) * inv * v[((size_t)b * S_ + key) * E_ + hh * HD_ + t];
    }
    o[((size_t)b * S_ + s) * E_ + hh * HD_ + t] = accv;
}

// ---------------- global attention for cls token (overwrites out row 0) ----------------
__global__ void lf_attn_global(const float* __restrict__ qg, const float* __restrict__ kg,
                               const float* __restrict__ vg, float* __restrict__ o) {
    int hh = blockIdx.x, b = blockIdx.y;
    int t = threadIdx.x;                                    // 256 threads; t == key index s
    __shared__ float p[S_];
    const float* qp = qg + (size_t)b * E_ + hh * HD_;
    const float* kp = kg + ((size_t)b * S_ + t) * E_ + hh * HD_;
    float d = 0.f;
    for (int i = 0; i < HD_; ++i) d += qp[i] * kp[i];
    float mx = blockMax(d);
    float e = expf(d - mx);
    float den = blockSum(e);
    p[t] = e / den;
    __syncthreads();
    if (t < HD_) {
        float accv = 0.f;
        for (int s2 = 0; s2 < S_; ++s2)
            accv += p[s2] * vg[((size_t)b * S_ + s2) * E_ + hh * HD_ + t];
        o[((size_t)b * S_) * E_ + hh * HD_ + t] = accv;
    }
}

// ---------------- residual add + LN (in place on h) ----------------
__global__ void lf_add_ln(float* __restrict__ h, const float* __restrict__ a,
                          const float* __restrict__ g, const float* __restrict__ bta, float eps) {
    int row = blockIdx.x, t = threadIdx.x;
    float* hr = h + (size_t)row * E_;
    const float* ar = a + (size_t)row * E_;
    float v0 = hr[t] + ar[t], v1 = hr[t + 256] + ar[t + 256], v2 = hr[t + 512] + ar[t + 512];
    float mean = blockSum(v0 + v1 + v2) * (1.0f / E_);
    float d0 = v0 - mean, d1 = v1 - mean, d2 = v2 - mean;
    float var = blockSum(d0 * d0 + d1 * d1 + d2 * d2) * (1.0f / E_);
    float inv = rsqrtf(var + eps);
    hr[t]       = d0 * inv * g[t]       + bta[t];
    hr[t + 256] = d1 * inv * g[t + 256] + bta[t + 256];
    hr[t + 512] = d2 * inv * g[t + 512] + bta[t + 512];
}

// ---------------- LN of h[:,0] -> pooled ----------------
__global__ void lf_rowln(const float* __restrict__ h, const float* __restrict__ g,
                         const float* __restrict__ bta, float* __restrict__ out, float eps) {
    int b = blockIdx.x, t = threadIdx.x;
    const float* hr = h + (size_t)b * S_ * E_;
    float v0 = hr[t], v1 = hr[t + 256], v2 = hr[t + 512];
    float mean = blockSum(v0 + v1 + v2) * (1.0f / E_);
    float d0 = v0 - mean, d1 = v1 - mean, d2 = v2 - mean;
    float var = blockSum(d0 * d0 + d1 * d1 + d2 * d2) * (1.0f / E_);
    float inv = rsqrtf(var + eps);
    float* o = out + (size_t)b * E_;
    o[t]       = d0 * inv * g[t]       + bta[t];
    o[t + 256] = d1 * inv * g[t + 256] + bta[t + 256];
    o[t + 512] = d2 * inv * g[t + 512] + bta[t + 512];
}

// ---------------- classifier: logits + softmax ----------------
__global__ void lf_logits(const float* __restrict__ z, const float* __restrict__ Wc,
                          const float* __restrict__ bc, float* __restrict__ out) {
    int b = blockIdx.x, t = threadIdx.x;
    __shared__ float lg[NCLS_];
    const float* zr = z + (size_t)b * MLPD_;
    for (int c = t; c < NCLS_; c += 256) {
        float accv = bc[c];
        for (int i = 0; i < MLPD_; ++i) accv += zr[i] * Wc[(size_t)i * NCLS_ + c];
        lg[c] = accv;
    }
    __syncthreads();
    float mx = -1e30f;
    for (int i = 0; i < NCLS_; ++i) mx = fmaxf(mx, lg[i]);
    float den = 0.f;
    for (int i = 0; i < NCLS_; ++i) den += expf(lg[i] - mx);
    for (int c = t; c < NCLS_; c += 256) out[(size_t)b * NCLS_ + c] = expf(lg[c] - mx) / den;
}

// ---------------- host orchestration ----------------
extern "C" void kernel_launch(void* const* d_in, const int* in_sizes, int n_in,
                              void* d_out, int out_size, void* d_ws, size_t ws_size,
                              hipStream_t stream) {
    (void)in_sizes; (void)n_in; (void)out_size; (void)ws_size;
    const float* x_feat  = (const float*)d_in[0];
    const int*   pos_ids = (const int*)d_in[1];
    const float* cls     = (const float*)d_in[2];
    const float* pos_emb = (const float*)d_in[3];
    const float* type_emb= (const float*)d_in[4];
    const float* emb_g   = (const float*)d_in[5];
    const float* emb_b   = (const float*)d_in[6];
    const float* bb_g    = (const float*)d_in[7];
    const float* bb_b    = (const float*)d_in[8];
    const float* Wq  = (const float*)d_in[9];
    const float* bq  = (const float*)d_in[10];
    const float* Wk  = (const float*)d_in[11];
    const float* bk  = (const float*)d_in[12];
    const float* Wv  = (const float*)d_in[13];
    const float* bv  = (const float*)d_in[14];
    const float* Wqg = (const float*)d_in[15];
    const float* bqg = (const float*)d_in[16];
    const float* Wkg = (const float*)d_in[17];
    const float* bkg = (const float*)d_in[18];
    const float* Wvg = (const float*)d_in[19];
    const float* bvg = (const float*)d_in[20];
    const float* Wo  = (const float*)d_in[21];
    const float* bo  = (const float*)d_in[22];
    const float* ln1g= (const float*)d_in[23];
    const float* ln1b= (const float*)d_in[24];
    const float* Wi  = (const float*)d_in[25];
    const float* bi  = (const float*)d_in[26];
    const float* Wo2 = (const float*)d_in[27];
    const float* bo2 = (const float*)d_in[28];
    const float* ln2g= (const float*)d_in[29];
    const float* ln2b= (const float*)d_in[30];
    const float* mlpg= (const float*)d_in[31];
    const float* mlpb= (const float*)d_in[32];
    const float* Wm  = (const float*)d_in[33];
    const float* bm  = (const float*)d_in[34];
    const float* Wc  = (const float*)d_in[35];
    const float* bc  = (const float*)d_in[36];

    char* wsp = (char*)d_ws;
    size_t off = 0;
    auto alloc = [&](size_t bytes) -> void* {
        off = (off + 255) & ~(size_t)255;
        void* p = wsp + off;
        off += bytes;
        return p;
    };
    const size_t BSE = (size_t)B_ * S_ * E_;
    float*          pooledHW = (float*)alloc((size_t)B_ * F_ * E_ * 4);
    float*          h    = (float*)alloc(BSE * 4);
    unsigned short* hbf  = (unsigned short*)alloc(BSE * 2);
    float*          q    = (float*)alloc(BSE * 4);
    float*          kbuf = (float*)alloc(BSE * 4);
    float*          vbuf = (float*)alloc(BSE * 4);
    float*          kg   = (float*)alloc(BSE * 4);
    float*          vg   = (float*)alloc(BSE * 4);
    float*          aout = (float*)alloc(BSE * 4);
    unsigned short* abf  = (unsigned short*)alloc(BSE * 2);
    float*          qg   = (float*)alloc((size_t)B_ * E_ * 4);
    unsigned short* ff1  = (unsigned short*)alloc((size_t)B_ * S_ * INTER_ * 2);
    unsigned short* wb   = (unsigned short*)alloc((size_t)E_ * INTER_ * 2);  // W^T bf16
    float*          pooled   = (float*)alloc((size_t)B_ * E_ * 4);
    unsigned short* pooledbf = (unsigned short*)alloc((size_t)B_ * E_ * 2);
    float*          zf   = (float*)alloc((size_t)B_ * MLPD_ * 4);

    auto cvt = [&](const float* src, unsigned short* dst, size_t n) {
        int blocks = (int)((n + 1023) / 1024);
        if (blocks > 2048) blocks = 2048;
        lf_cvt<<<blocks, 256, 0, stream>>>(src, dst, n);
    };
    auto cvtT = [&](const float* src, unsigned short* dst, int K, int N) {
        size_t n = (size_t)K * N;
        int blocks = (int)((n + 1023) / 1024);
        if (blocks > 2048) blocks = 2048;
        lf_cvtT<<<blocks, 256, 0, stream>>>(src, dst, K, N);
    };
    // BT is transposed weights [N,K]; ldbt == K
    auto gemm = [&](const void* Ab, int lda, const void* BTb, int ldbt, void* C, int ldc,
                    const float* bias, int M, int N, int K, float alpha, int act, int outbf) {
        dim3 g(N / 64, (M + 127) / 128);
        lf_gemm<<<g, 256, 0, stream>>>((const bf16*)Ab, lda, (const bf16*)BTb, ldbt,
                                       C, ldc, bias, M, N, K, alpha, act, outbf);
    };

    lf_lnmax<<<B_ * F_, 256, 0, stream>>>(x_feat, bb_g, bb_b, pooledHW);
    lf_embed<<<B_ * S_, 256, 0, stream>>>(pooledHW, pos_ids, cls, pos_emb, type_emb,
                                          emb_g, emb_b, h);

    const float scale = 0.125f;     // 1/sqrt(64)
    const int MS = B_ * S_;
    for (int l = 0; l < NL_; ++l) {
        const size_t wEE = (size_t)l * E_ * E_;
        cvt(h, hbf, BSE);
        cvtT(Wq + wEE, wb, E_, E_);
        gemm(hbf, E_, wb, E_, q, E_, bq + l * E_, MS, E_, E_, scale, 0, 0);
        cvtT(Wk + wEE, wb, E_, E_);
        gemm(hbf, E_, wb, E_, kbuf, E_, bk + l * E_, MS, E_, E_, 1.0f, 0, 0);
        cvtT(Wv + wEE, wb, E_, E_);
        gemm(hbf, E_, wb, E_, vbuf, E_, bv + l * E_, MS, E_, E_, 1.0f, 0, 0);
        cvtT(Wkg + wEE, wb, E_, E_);
        gemm(hbf, E_, wb, E_, kg, E_, bkg + l * E_, MS, E_, E_, 1.0f, 0, 0);
        cvtT(Wvg + wEE, wb, E_, E_);
        gemm(hbf, E_, wb, E_, vg, E_, bvg + l * E_, MS, E_, E_, 1.0f, 0, 0);
        // qg from h[:,0] rows only: A rows at stride S*E, M = B
        cvtT(Wqg + wEE, wb, E_, E_);
        gemm(hbf, S_ * E_, wb, E_, qg, E_, bqg + l * E_, B_, E_, E_, scale, 0, 0);

        lf_attn_local<<<dim3(S_, NH_, B_), 64, 0, stream>>>(q, kbuf, vbuf, aout);
        lf_attn_global<<<dim3(NH_, B_), 256, 0, stream>>>(qg, kg, vg, aout);

        cvt(aout, abf, BSE);
        cvtT(Wo + wEE, wb, E_, E_);
        gemm(abf, E_, wb, E_, q, E_, bo + l * E_, MS, E_, E_, 1.0f, 0, 0);
        lf_add_ln<<<MS, 256, 0, stream>>>(h, q, ln1g + l * E_, ln1b + l * E_, 1e-5f);

        cvt(h, hbf, BSE);
        cvtT(Wi + (size_t)l * E_ * INTER_, wb, E_, INTER_);
        gemm(hbf, E_, wb, E_, ff1, INTER_, bi + l * INTER_, MS, INTER_, E_, 1.0f, 1, 1);
        cvtT(Wo2 + (size_t)l * INTER_ * E_, wb, INTER_, E_);
        gemm(ff1, INTER_, wb, INTER_, q, E_, bo2 + l * E_, MS, E_, INTER_, 1.0f, 0, 0);
        lf_add_ln<<<MS, 256, 0, stream>>>(h, q, ln2g + l * E_, ln2b + l * E_, 1e-5f);
    }

    lf_rowln<<<B_, 256, 0, stream>>>(h, mlpg, mlpb, pooled, 1e-3f);
    cvt(pooled, pooledbf, (size_t)B_ * E_);
    cvtT(Wm, wb, E_, MLPD_);
    gemm(pooledbf, E_, wb, E_, zf, MLPD_, bm, B_, MLPD_, E_, 1.0f, 1, 0);
    lf_logits<<<B_, 256, 0, stream>>>(zf, Wc, bc, (float*)d_out);
}
